// OPTAttention_59777354826050
// MI455X (gfx1250) — matrix operand order repaired and checked by reading, not compiled, not run on hardware
//
#include <hip/hip_runtime.h>

typedef __attribute__((ext_vector_type(16))) __bf16 v16bf;
typedef __attribute__((ext_vector_type(8)))  __bf16 v8bf;
typedef __attribute__((ext_vector_type(8)))  float  v8f;

#define SEQ   2048
#define DIM   2048
#define NHEAD 32
#define HD    64
#define N3    6144
#define SCALE 0.125f

// ---------------------------------------------------------------------------
// WMMA wrapper: D = A(16x32 bf16) x B(32x16 bf16) + C(16x16 f32)
// ---------------------------------------------------------------------------
__device__ __forceinline__ v8f wmma_bf16(v16bf a, v16bf b, v8f c) {
  return __builtin_amdgcn_wmma_f32_16x16x32_bf16(
      /*neg_a=*/false, a, /*neg_b=*/false, b,
      /*c_mod=*/(short)0, c, /*reuse_a=*/false, /*reuse_b=*/false);
}

// A-matrix (16x32, M x K) fragment from fp32 row-major [*, ld]:
// lane (half=lane>>4, m=lane&15): elems 0..15 = K[k0+16h .. +15]
__device__ __forceinline__ v16bf load_a_f32(const float* p, int ld, int row0,
                                            int k0, int m, int half) {
  const float* r = p + (size_t)(row0 + m) * ld + k0 + 16 * half;
  v16bf f;
#pragma unroll
  for (int i = 0; i < 8; ++i) f[i] = (__bf16)r[i];
#pragma unroll
  for (int i = 0; i < 8; ++i) f[8 + i] = (__bf16)r[8 + i];
  return f;
}

// Same A layout from bf16 row-major (16B-aligned chunks)
__device__ __forceinline__ v16bf load_a_bf16(const __bf16* p, int ld, int row0,
                                             int k0, int m, int half) {
  const __bf16* r = p + (size_t)(row0 + m) * ld + k0 + 16 * half;
  v8bf lo = *(const v8bf*)(r);
  v8bf hi = *(const v8bf*)(r + 8);
  return __builtin_shufflevector(lo, hi, 0, 1, 2, 3, 4, 5, 6, 7,
                                 8, 9, 10, 11, 12, 13, 14, 15);
}

// B-matrix (32x16, K x N) fragment where column n is a contiguous K-run in
// memory (i.e. B = Wt with W row-major [N,K]):
// lane (half, n): elems 0..15 = K[k0+16h .. +15] of column n
__device__ __forceinline__ v16bf load_b_f32(const float* p, int ld, int col0,
                                            int k0, int n, int half) {
  const float* r = p + (size_t)(col0 + n) * ld + k0 + 16 * half;
  v16bf f;
#pragma unroll
  for (int i = 0; i < 16; ++i) f[i] = (__bf16)r[i];
  return f;
}

__device__ __forceinline__ v16bf load_b_bf16(const __bf16* p, size_t ld,
                                             int col0, int k0, int n, int half) {
  const __bf16* r = p + (size_t)(col0 + n) * ld + k0 + 16 * half;
  return *(const v16bf*)r;  // 32B aligned by construction
}

// ---------------------------------------------------------------------------
// Kernel 1: fused QKV projection.  X[2048,2048]f32 x Wqkv[6144,2048]f32^T + b
// -> Q,K as bf16 [H][S][64], V as bf16 transposed [H][64][S].
// 4 waves/block, each wave owns a 64x64 C tile, block tile 128x128.
// ---------------------------------------------------------------------------
__global__ __launch_bounds__(128) void qkv_gemm_kernel(
    const float* __restrict__ X, const float* __restrict__ W,
    const float* __restrict__ bias, __bf16* __restrict__ Qb,
    __bf16* __restrict__ Kb, __bf16* __restrict__ Vt) {
  const int wid = threadIdx.x >> 5;
  const int lane = threadIdx.x & 31;
  const int half = lane >> 4;
  const int ln = lane & 15;
  const int m0 = blockIdx.y * 128 + (wid >> 1) * 64;
  const int n0 = blockIdx.x * 128 + (wid & 1) * 64;

  v8f acc[4][4] = {};

  for (int k0 = 0; k0 < DIM; k0 += 32) {
    v16bf a[4], b[4];
#pragma unroll
    for (int mt = 0; mt < 4; ++mt)
      a[mt] = load_a_f32(X, DIM, m0 + mt * 16, k0, ln, half);
#pragma unroll
    for (int nt = 0; nt < 4; ++nt)
      b[nt] = load_b_f32(W, DIM, n0 + nt * 16, k0, ln, half);
#pragma unroll
    for (int mt = 0; mt < 4; ++mt)
#pragma unroll
      for (int nt = 0; nt < 4; ++nt)
        acc[mt][nt] = wmma_bf16(a[mt], b[nt], acc[mt][nt]);
  }

  // Epilogue: +bias, route to Q / K / V(transposed) in bf16.
#pragma unroll
  for (int nt = 0; nt < 4; ++nt) {
    const int e = n0 + nt * 16 + ln;       // output channel in [0, 6144)
    const float bv = bias[e];
    const int sec = e >> 11;               // 0=Q 1=K 2=V
    const int ei = e & 2047;
    const int h = ei >> 6;
    const int dd = ei & 63;
#pragma unroll
    for (int mt = 0; mt < 4; ++mt) {
#pragma unroll
      for (int r = 0; r < 8; ++r) {
        const int s = m0 + mt * 16 + r + 8 * half;
        const __bf16 v = (__bf16)(acc[mt][nt][r] + bv);
        if (sec == 0)
          Qb[((size_t)h * SEQ + s) * HD + dd] = v;
        else if (sec == 1)
          Kb[((size_t)h * SEQ + s) * HD + dd] = v;
        else
          Vt[((size_t)h * HD + dd) * SEQ + s] = v;
      }
    }
  }
}

// ---------------------------------------------------------------------------
// Kernel 2: causal flash attention.  grid = (S/64, H), 4 waves/block,
// each wave owns 16 query rows; kv loop trip count uniform per block.
// ---------------------------------------------------------------------------
__global__ __launch_bounds__(128) void flash_attn_kernel(
    const __bf16* __restrict__ Q, const __bf16* __restrict__ K,
    const __bf16* __restrict__ Vt, __bf16* __restrict__ Ab) {
  // per-wave P tile (row-major 16x64); 32B-aligned for v8bf/b128 LDS access
  __shared__ __attribute__((aligned(32))) __bf16 plds[4][16 * 64];

  const int wid = threadIdx.x >> 5;
  const int lane = threadIdx.x & 31;
  const int half = lane >> 4;
  const int ln = lane & 15;
  const int h = blockIdx.y;
  const int qblk = blockIdx.x;
  const int qbase = qblk * 64 + wid * 16;  // this wave's first query row

  const __bf16* Qh = Q + (size_t)h * SEQ * HD;
  const __bf16* Kh = K + (size_t)h * SEQ * HD;
  const __bf16* Vh = Vt + (size_t)h * HD * SEQ;

  // Q fragments for this wave's 16 rows (Hd = 64 -> 2 k-steps of 32)
  v16bf qa[2];
#pragma unroll
  for (int kk = 0; kk < 2; ++kk)
    qa[kk] = load_a_bf16(Qh, HD, qbase, kk * 32, ln, half);

  v8f acco[4] = {};
  float mrow[8], lrow[8];
#pragma unroll
  for (int r = 0; r < 8; ++r) { mrow[r] = -3.0e38f; lrow[r] = 0.0f; }

  const int nblk = qblk + 1;
  for (int jb = 0; jb < nblk; ++jb) {
    const int j0 = jb * 64;

    // ---- S = Q x K^T (16x64 scores in 4 C-tiles) ----
    v8f sc[4] = {};
#pragma unroll
    for (int t = 0; t < 4; ++t)
#pragma unroll
      for (int kk = 0; kk < 2; ++kk) {
        v16bf kb = load_b_bf16(Kh, HD, j0 + t * 16, kk * 32, ln, half);
        sc[t] = wmma_bf16(qa[kk], kb, sc[t]);
      }

    const bool edge = (j0 + 63 > qbase);  // wave-uniform causal edge block

    // ---- online softmax, row = r + 8*half, col = j0 + t*16 + ln ----
#pragma unroll
    for (int r = 0; r < 8; ++r) {
      const int qq = qbase + r + 8 * half;
      float mx = -3.0e38f;
#pragma unroll
      for (int t = 0; t < 4; ++t) {
        float v = sc[t][r] * SCALE;
        if (edge && (j0 + t * 16 + ln) > qq) v = -3.0e38f;
        sc[t][r] = v;
        mx = fmaxf(mx, v);
      }
#pragma unroll
      for (int off = 8; off >= 1; off >>= 1)
        mx = fmaxf(mx, __shfl_xor(mx, off, 32));
      const float mnew = fmaxf(mrow[r], mx);
      const float alpha = __expf(mrow[r] - mnew);
      mrow[r] = mnew;
      float rs = 0.0f;
#pragma unroll
      for (int t = 0; t < 4; ++t) {
        const float p = __expf(sc[t][r] - mnew);
        sc[t][r] = p;
        rs += p;
      }
#pragma unroll
      for (int off = 8; off >= 1; off >>= 1) rs += __shfl_xor(rs, off, 32);
      lrow[r] = lrow[r] * alpha + rs;
#pragma unroll
      for (int t = 0; t < 4; ++t) acco[t][r] *= alpha;
    }

    // ---- C-layout -> A-layout for P via per-wave LDS tile ----
    __bf16* pl = plds[wid];
#pragma unroll
    for (int t = 0; t < 4; ++t)
#pragma unroll
      for (int r = 0; r < 8; ++r)
        pl[(r + 8 * half) * 64 + t * 16 + ln] = (__bf16)sc[t][r];
    asm volatile("s_wait_dscnt 0" ::: "memory");

    // ---- O += P(16x64) x V(64x64), V pre-transposed so B loads contiguous
#pragma unroll
    for (int kk = 0; kk < 2; ++kk) {
      v16bf pa = load_a_bf16(pl, 64, 0, kk * 32, ln, half);
#pragma unroll
      for (int t = 0; t < 4; ++t) {
        v16bf vb = load_b_bf16(Vh + j0, SEQ, t * 16, kk * 32, ln, half);
        acco[t] = wmma_bf16(pa, vb, acco[t]);
      }
    }
  }

  // ---- normalize and write attn output bf16 [S][D], D idx = h*64 + d ----
#pragma unroll
  for (int r = 0; r < 8; ++r) {
    const float inv = 1.0f / lrow[r];
    const int qq = qbase + r + 8 * half;
#pragma unroll
    for (int t = 0; t < 4; ++t)
      Ab[(size_t)qq * DIM + h * 64 + t * 16 + ln] = (__bf16)(acco[t][r] * inv);
  }
}

// ---------------------------------------------------------------------------
// Kernel 3: output projection.  attn[2048,2048]bf16 x Wout[2048,2048]f32^T + b
// ---------------------------------------------------------------------------
__global__ __launch_bounds__(128) void out_gemm_kernel(
    const __bf16* __restrict__ A, const float* __restrict__ W,
    const float* __restrict__ bias, float* __restrict__ Out) {
  const int wid = threadIdx.x >> 5;
  const int lane = threadIdx.x & 31;
  const int half = lane >> 4;
  const int ln = lane & 15;
  const int m0 = blockIdx.y * 128 + (wid >> 1) * 64;
  const int n0 = blockIdx.x * 128 + (wid & 1) * 64;

  v8f acc[4][4] = {};

  for (int k0 = 0; k0 < DIM; k0 += 32) {
    v16bf a[4], b[4];
#pragma unroll
    for (int mt = 0; mt < 4; ++mt)
      a[mt] = load_a_bf16(A, DIM, m0 + mt * 16, k0, ln, half);
#pragma unroll
    for (int nt = 0; nt < 4; ++nt)
      b[nt] = load_b_f32(W, DIM, n0 + nt * 16, k0, ln, half);
#pragma unroll
    for (int mt = 0; mt < 4; ++mt)
#pragma unroll
      for (int nt = 0; nt < 4; ++nt)
        acc[mt][nt] = wmma_bf16(a[mt], b[nt], acc[mt][nt]);
  }

#pragma unroll
  for (int nt = 0; nt < 4; ++nt) {
    const int e = n0 + nt * 16 + ln;
    const float bv = bias[e];
#pragma unroll
    for (int mt = 0; mt < 4; ++mt)
#pragma unroll
      for (int r = 0; r < 8; ++r) {
        const int s = m0 + mt * 16 + r + 8 * half;
        Out[(size_t)s * DIM + e] = acc[mt][nt][r] + bv;
      }
  }
}

// ---------------------------------------------------------------------------
extern "C" void kernel_launch(void* const* d_in, const int* in_sizes, int n_in,
                              void* d_out, int out_size, void* d_ws,
                              size_t ws_size, hipStream_t stream) {
  const float* X    = (const float*)d_in[0];  // [1,2048,2048]
  const float* Wqkv = (const float*)d_in[1];  // [6144,2048]
  const float* bqkv = (const float*)d_in[2];  // [6144]
  const float* Wout = (const float*)d_in[3];  // [2048,2048]
  const float* bout = (const float*)d_in[4];  // [2048]
  float* Out = (float*)d_out;

  const size_t HSD = (size_t)NHEAD * SEQ * HD;  // 4M elems
  __bf16* Qb = (__bf16*)d_ws;
  __bf16* Kb = Qb + HSD;
  __bf16* Vt = Kb + HSD;
  __bf16* Ab = Vt + HSD;  // total 16M bf16 = 32 MiB

  qkv_gemm_kernel<<<dim3(N3 / 128, SEQ / 128), 128, 0, stream>>>(
      X, Wqkv, bqkv, Qb, Kb, Vt);
  flash_attn_kernel<<<dim3(SEQ / 64, NHEAD), 128, 0, stream>>>(Qb, Kb, Vt, Ab);
  out_gemm_kernel<<<dim3(DIM / 128, SEQ / 128), 128, 0, stream>>>(
      Ab, Wout, bout, Out);
}
